// NetworkLayer_79173427134941
// MI455X (gfx1250) — matrix operand order repaired and checked by reading, not compiled, not run on hardware
//
#include <hip/hip_runtime.h>
#include <stdint.h>

typedef __attribute__((ext_vector_type(16))) _Float16 v16h;
typedef __attribute__((ext_vector_type(8)))  float    v8f;

#define B_B   32
#define N_N   1024
#define G_G   8
#define H_H   128
#define DIN_  1033          // 8 (u) + 1 (norm) + 1024 (dots)
#define K1_CHUNKS 33        // ceil(1033/32)
#define K2_CHUNKS 4         // 128/32
#define MTILE 128
#define ROWBLOCKS (N_N / MTILE)   // 8

// workspace layout in uint32 words: W0/W1/W2 pre-swizzled into WMMA B-fragment order
#define W0SW_WORDS (K1_CHUNKS*8*32*8)             // 67584
#define W1SW_OFF   W0SW_WORDS
#define W1SW_WORDS (K2_CHUNKS*8*32*8)             // 8192
#define W2SW_OFF   (W1SW_OFF + W1SW_WORDS)        // 75776
#define TOTAL_SW_WORDS (W2SW_OFF + W1SW_WORDS)    // 83968 words = 328 KB
#define OUT_ELEMS  (B_B*H_H*3)                    // 12288

static __device__ inline uint32_t pack2h(float a, float b) {
  _Float16 ha = (_Float16)a, hb = (_Float16)b;
  uint16_t ua, ub;
  __builtin_memcpy(&ua, &ha, 2);
  __builtin_memcpy(&ub, &hb, 2);
  return (uint32_t)ua | ((uint32_t)ub << 16);
}

// raw v_sqrt_f32 (inputs are small non-negative dots; no IEEE fixup needed)
static __device__ inline float fsqrt(float v) { return __builtin_amdgcn_sqrtf(v); }
static __device__ inline float dot3(float4 a, float4 b) {
  return a.x*b.x + a.y*b.y + a.z*b.z;
}

// ---------------------------------------------------------------------------
// Prep: zero d_out; convert W0/W1/W2 f32 -> f16 packed in exact B-fragment
// register order: word index = ((chunk*8 + tile)*32 + lane)*8 + vgpr,
// element K (per 16-bit B layout) = 2*vgpr + half + (vgpr>=4 ? 8 : 0) + (lane>=16 ? 8 : 0),
// column n = (lane&15) + 16*tile.  Rows beyond K-dim are zero padded.
// ---------------------------------------------------------------------------
__global__ void prep_kernel(const float* __restrict__ W0,
                            const float* __restrict__ W1,
                            const float* __restrict__ W2,
                            uint32_t* __restrict__ wsw,
                            float* __restrict__ out) {
  int gid = blockIdx.x * blockDim.x + threadIdx.x;
  if (gid < OUT_ELEMS) out[gid] = 0.0f;
  if (gid >= TOTAL_SW_WORDS) return;

  const float* src; int kmax; int idx;
  if (gid < W0SW_WORDS)      { src = W0; kmax = DIN_; idx = gid; }
  else if (gid < W2SW_OFF)   { src = W1; kmax = H_H;  idx = gid - W1SW_OFF; }
  else                       { src = W2; kmax = H_H;  idx = gid - W2SW_OFF; }

  int v    =  idx        & 7;
  int lane = (idx >> 3)  & 31;
  int tile = (idx >> 8)  & 7;
  int chunk =  idx >> 11;

  int k0 = chunk*32 + 2*v + ((v >= 4) ? 8 : 0) + ((lane >= 16) ? 8 : 0);
  int n  = (lane & 15) + tile*16;
  float f0 = (k0     < kmax) ? src[(size_t)k0     * H_H + n] : 0.0f;
  float f1 = (k0 + 1 < kmax) ? src[(size_t)(k0+1) * H_H + n] : 0.0f;
  wsw[gid] = pack2h(f0, f1);
}

// ---------------------------------------------------------------------------
// Fused kernel: one workgroup = (batch b, 128-row block). 8 waves; wave w owns
// rows w*16..w*16+15 and all 128 MLP columns.
// ---------------------------------------------------------------------------
__global__ __launch_bounds__(256)
void fused_kernel(const float* __restrict__ x, const float* __restrict__ u,
                  const float* __restrict__ b0, const float* __restrict__ b1,
                  const float* __restrict__ b2,
                  const uint32_t* __restrict__ wsw,
                  float* __restrict__ out) {
  __shared__ float4   xall[N_N];          // all 1024 points of this batch (16 KB)
  __shared__ _Float16 hbuf[MTILE][H_H];   // inter-layer activations (32 KB)
  __shared__ float    ubuf[G_G];

  const int b    = blockIdx.x >> 3;
  const int blk  = blockIdx.x & 7;
  const int row0 = blk * MTILE;
  const int tid  = threadIdx.x;
  const int wave = tid >> 5;
  const int lane = tid & 31;

  const float* xb = x + (size_t)b * N_N * 3;
  for (int r = tid; r < N_N; r += 256)
    xall[r] = make_float4(xb[r*3+0], xb[r*3+1], xb[r*3+2], 0.0f);
  if (tid < G_G) ubuf[tid] = u[b*G_G + tid];
  __syncthreads();

  const int m         = lane & 15;         // A-fragment row within 16-row tile
  const int row_local = wave*16 + m;
  const float4 xi     = xall[row0 + row_local];
  const float norm_i  = fsqrt(xi.x*xi.x + xi.y*xi.y + xi.z*xi.z);
  const int klo       = (lane >= 16) ? 8 : 0;   // A-layout lane K offset
  const v16h* wb      = (const v16h*)wsw;

  v8f acc[8];
  #pragma unroll
  for (int t = 0; t < 8; ++t)
    #pragma unroll
    for (int r = 0; r < 8; ++r) acc[t][r] = 0.0f;

  // ---------------- GEMM1: scalars[1024 x 1033] @ W0 (A built on the fly) ---
  // chunk 0: mixed u / norm / dots (branchless selects; only chunk with them)
  {
    v16h a;
    #pragma unroll
    for (int ei = 0; ei < 16; ++ei) {
      int kg = ei + ((ei >= 8) ? 8 : 0) + klo;  // 0..31
      int j  = (kg >= 9) ? (kg - 9) : 0;        // clamp to valid LDS index
      float dv  = fsqrt(dot3(xi, xall[j]));
      float val = (kg < G_G) ? ubuf[kg & 7] : ((kg == G_G) ? norm_i : dv);
      a[ei] = (_Float16)val;
    }
    #pragma unroll
    for (int t = 0; t < 8; ++t) {
      v16h bf = wb[(size_t)t*32 + lane];
      acc[t] = __builtin_amdgcn_wmma_f32_16x16x32_f16(
          false, a, false, bf, (short)0, acc[t], false, false);
    }
  }
  // chunks 1..31: pure dots, zero conditionals in the hot loop
  for (int c = 1; c < 32; ++c) {
    v16h a;
    #pragma unroll
    for (int ei = 0; ei < 16; ++ei) {
      int j = c*32 + ei + ((ei >= 8) ? 8 : 0) + klo - 9;   // 23..1014 < N_N
      a[ei] = (_Float16)fsqrt(dot3(xi, xall[j]));
    }
    #pragma unroll
    for (int t = 0; t < 8; ++t) {
      v16h bf = wb[(size_t)(c*8 + t)*32 + lane];
      acc[t] = __builtin_amdgcn_wmma_f32_16x16x32_f16(
          false, a, false, bf, (short)0, acc[t], false, false);
    }
  }
  // chunk 32: tail with j < N guard
  {
    const int c = 32;
    v16h a;
    #pragma unroll
    for (int ei = 0; ei < 16; ++ei) {
      int j  = c*32 + ei + ((ei >= 8) ? 8 : 0) + klo - 9;
      int jc = (j < N_N) ? j : 0;
      float dv = fsqrt(dot3(xi, xall[jc]));
      a[ei] = (_Float16)((j < N_N) ? dv : 0.0f);
    }
    #pragma unroll
    for (int t = 0; t < 8; ++t) {
      v16h bf = wb[(size_t)(c*8 + t)*32 + lane];
      acc[t] = __builtin_amdgcn_wmma_f32_16x16x32_f16(
          false, a, false, bf, (short)0, acc[t], false, false);
    }
  }

  // epilogue 1: + b0, leaky_relu, -> hbuf (f16)
  const int crow  = wave*16 + ((lane >= 16) ? 8 : 0);  // C layout M base
  const int ccol0 = lane & 15;
  #pragma unroll
  for (int t = 0; t < 8; ++t) {
    int col = ccol0 + t*16;
    float bb = b0[col];
    #pragma unroll
    for (int r = 0; r < 8; ++r) {
      float o = acc[t][r] + bb;
      o = (o > 0.0f) ? o : 0.01f * o;
      hbuf[crow + r][col] = (_Float16)o;
    }
  }
  __syncthreads();

  // ---------------- GEMM2: h @ W1 ------------------------------------------
  #pragma unroll
  for (int t = 0; t < 8; ++t)
    #pragma unroll
    for (int r = 0; r < 8; ++r) acc[t][r] = 0.0f;
  for (int c = 0; c < K2_CHUNKS; ++c) {
    v16h a;
    #pragma unroll
    for (int ei = 0; ei < 16; ++ei) {
      int k = c*32 + ei + ((ei >= 8) ? 8 : 0) + klo;
      a[ei] = hbuf[row_local][k];
    }
    #pragma unroll
    for (int t = 0; t < 8; ++t) {
      v16h bf = wb[(size_t)(W1SW_OFF/8) + (size_t)(c*8 + t)*32 + lane];
      acc[t] = __builtin_amdgcn_wmma_f32_16x16x32_f16(
          false, a, false, bf, (short)0, acc[t], false, false);
    }
  }
  __syncthreads();
  #pragma unroll
  for (int t = 0; t < 8; ++t) {
    int col = ccol0 + t*16;
    float bb = b1[col];
    #pragma unroll
    for (int r = 0; r < 8; ++r) {
      float o = acc[t][r] + bb;
      o = (o > 0.0f) ? o : 0.01f * o;
      hbuf[crow + r][col] = (_Float16)o;
    }
  }
  __syncthreads();

  // ---------------- GEMM3: h @ W2 ------------------------------------------
  #pragma unroll
  for (int t = 0; t < 8; ++t)
    #pragma unroll
    for (int r = 0; r < 8; ++r) acc[t][r] = 0.0f;
  for (int c = 0; c < K2_CHUNKS; ++c) {
    v16h a;
    #pragma unroll
    for (int ei = 0; ei < 16; ++ei) {
      int k = c*32 + ei + ((ei >= 8) ? 8 : 0) + klo;
      a[ei] = hbuf[row_local][k];
    }
    #pragma unroll
    for (int t = 0; t < 8; ++t) {
      v16h bf = wb[(size_t)(W2SW_OFF/8) + (size_t)(c*8 + t)*32 + lane];
      acc[t] = __builtin_amdgcn_wmma_f32_16x16x32_f16(
          false, a, false, bf, (short)0, acc[t], false, false);
    }
  }
  __syncthreads();
  // fk = acc + b2 -> hbuf (f16)
  #pragma unroll
  for (int t = 0; t < 8; ++t) {
    int col = ccol0 + t*16;
    float bb = b2[col];
    #pragma unroll
    for (int r = 0; r < 8; ++r)
      hbuf[crow + r][col] = (_Float16)(acc[t][r] + bb);
  }
  __syncthreads();

  // ---------------- out[b,o,:] += sum_i fk[i,o] * x[i,:] / N ----------------
  {
    int o    = tid & 127;
    int half = tid >> 7;
    float s0 = 0.0f, s1 = 0.0f, s2 = 0.0f;
    for (int i = half*64; i < half*64 + 64; ++i) {
      float f = (float)hbuf[i][o];
      float4 xr = xall[row0 + i];
      s0 += f * xr.x; s1 += f * xr.y; s2 += f * xr.z;
    }
    const float scale = 1.0f / (float)N_N;
    float* op = out + ((size_t)b*H_H + o)*3;
    atomicAdd(op + 0, s0 * scale);
    atomicAdd(op + 1, s1 * scale);
    atomicAdd(op + 2, s2 * scale);
  }
}

extern "C" void kernel_launch(void* const* d_in, const int* in_sizes, int n_in,
                              void* d_out, int out_size, void* d_ws, size_t ws_size,
                              hipStream_t stream) {
  (void)in_sizes; (void)n_in; (void)out_size; (void)ws_size;
  const float* x  = (const float*)d_in[0];
  const float* u  = (const float*)d_in[1];
  const float* W0 = (const float*)d_in[2];
  const float* b0 = (const float*)d_in[3];
  const float* W1 = (const float*)d_in[4];
  const float* b1 = (const float*)d_in[5];
  const float* W2 = (const float*)d_in[6];
  const float* b2 = (const float*)d_in[7];
  float*    out = (float*)d_out;
  uint32_t* wsw = (uint32_t*)d_ws;

  // swizzle weights to f16 fragments + zero the output accumulator
  prep_kernel<<<(TOTAL_SW_WORDS + 255)/256, 256, 0, stream>>>(W0, W1, W2, wsw, out);
  // fused dots + 3-layer MLP + contraction
  fused_kernel<<<B_B * ROWBLOCKS, 256, 0, stream>>>(x, u, b0, b1, b2, wsw, out);
}
